// FilterLayer_39402029973899
// MI455X (gfx1250) — compile-verified, matrix operand order checked by reading, not run on hardware
//
#include <hip/hip_runtime.h>

// ---------------------------------------------------------------------------
// FilterLayer (FNet-style frequency filter) for MI455X / gfx1250.
//
// irfft(w * rfft(x)) == Ti @ (w .* (Tf @ x)) with channel-shared real
// transform matrices Tf (rfft: cos/sin rows) and Ti (irfft with half-spectrum
// 2x factors). Two bf16 WMMA GEMMs per (stream,batch) row, fused with the
// embedding gather, residual add and LayerNorm. Memory-bound (~390MB moved,
// ~17us at 23.3 TB/s); WMMA keeps the 26 GFLOP of transform math off the
// critical path.
// ---------------------------------------------------------------------------

#define SEQ   200   // sequence length
#define EMB   128   // embedding dim
#define SP    224   // padded seq/freq dim (14 tiles of 16, 7 K-tiles of 32)
#define FROW  132   // padded f32 LDS row (breaks 128-stride bank conflicts)
#define NFREQ 101   // rfft bins

// LDS layout (bytes):
//   xf  f32 [200][132]  @ 0        size 105600   (residual source)
//   xt  bf16[128][224]  @ 105600   size  57344   (B operand; later G)
//   Fb  f32 [224][132]  @ 162944   size 118272   (F spectrum; later h)
#define XT_OFF 105600
#define FB_OFF 162944
#define LDS_BYTES 281216

typedef __attribute__((ext_vector_type(16))) __bf16 v16bf;
typedef __attribute__((ext_vector_type(8)))  __bf16 bf16x8;
typedef __attribute__((ext_vector_type(8)))  float  v8f;

// -------------------------- transform build --------------------------------
// Tf rows:  k in [0,100]    : cos(2*pi*k*s/200)/sqrt(200)
//           k in [112,212]  : -sin(2*pi*(k-112)*s/200)/sqrt(200)   (Fi rows)
// Ti cols:  0: 1/sqrt(200); 1..99: 2cos/sqrt; 100: (-1)^t/sqrt;
//           113..211: -2sin/sqrt; cols 112 and 212 are zero (imag parts of
//           DC/Nyquist are ignored by irfft). Everything else zero padding.
__global__ void build_transforms(__bf16* __restrict__ Tf, __bf16* __restrict__ Ti) {
    const int i = blockIdx.x * blockDim.x + threadIdx.x;
    const int N2 = SP * SP;
    if (i >= 2 * N2) return;
    const float inv = 0.07071067811865475f;          // 1/sqrt(200)
    const float w0  = 6.283185307179586f / 200.f;    // 2*pi/200
    if (i < N2) {
        const int k = i / SP, s = i % SP;
        float v = 0.f;
        if (s < SEQ) {
            if (k <= 100)
                v = cosf(w0 * (float)((k * s) % 200)) * inv;
            else if (k >= 112 && k <= 212)
                v = -sinf(w0 * (float)(((k - 112) * s) % 200)) * inv;
        }
        Tf[i] = (__bf16)v;
    } else {
        const int j = i - N2;
        const int t = j / SP, kc = j % SP;
        float v = 0.f;
        if (t < SEQ) {
            if (kc == 0)
                v = inv;
            else if (kc <= 99)
                v = 2.f * cosf(w0 * (float)((kc * t) % 200)) * inv;
            else if (kc == 100)
                v = (t & 1) ? -inv : inv;
            else if (kc >= 113 && kc <= 211)
                v = -2.f * sinf(w0 * (float)(((kc - 112) * t) % 200)) * inv;
        }
        Ti[j] = (__bf16)v;
    }
}

// ------------------------------ fragments ----------------------------------
// 16-bit A 16x32 layout (ISA 7.12.2): lane<16 -> M=lane, elems 0..7 = K 0..7,
// elems 8..15 = K 16..23; lane>=16 -> same M rows, K base +8. B 32x16 mirrors
// with lane = N. Both reduce to: per lane, two contiguous 8xbf16 (16B) chunks
// at col (kbase) and (kbase+16), row = (M or N).
__device__ __forceinline__ v16bf load_frag(const __bf16* p) {
    const bf16x8* q = (const bf16x8*)p;
    bf16x8 lo = q[0];
    bf16x8 hi = q[2];   // +16 bf16
    return __builtin_shufflevector(lo, hi, 0, 1, 2, 3, 4, 5, 6, 7,
                                           8, 9, 10, 11, 12, 13, 14, 15);
}

// 7 waves; wave w owns M-tiles {w, w+7} of 14; all 8 E(N)-tiles; K = 7x32.
__device__ __forceinline__ void run_gemm(const __bf16* __restrict__ A,
                                         const __bf16* __restrict__ B,
                                         int wid, int lane, v8f acc[2][8]) {
    const int row = lane & 15;
    const int kb  = (lane >> 4) * 8;
    for (int k = 0; k < 7; ++k) {
        const int kcol = k * 32 + kb;
        v16bf af0 = load_frag(A + ((wid * 16 + row) * SP + kcol));
        v16bf af1 = load_frag(A + (((wid + 7) * 16 + row) * SP + kcol));
#pragma unroll
        for (int e8 = 0; e8 < 8; ++e8) {
            v16bf bf = load_frag(B + ((e8 * 16 + row) * SP + kcol));
            acc[0][e8] = __builtin_amdgcn_wmma_f32_16x16x32_bf16(
                false, af0, false, bf, (short)0, acc[0][e8], false, false);
            acc[1][e8] = __builtin_amdgcn_wmma_f32_16x16x32_bf16(
                false, af1, false, bf, (short)0, acc[1][e8], false, false);
        }
    }
}

// ------------------------------ main kernel --------------------------------
__global__ __launch_bounds__(224, 1)
void filter_fused(const int* __restrict__ qids, const int* __restrict__ uids,
                  const int* __restrict__ vids, const int* __restrict__ clicks,
                  const float* __restrict__ qtab, const float* __restrict__ utab,
                  const float* __restrict__ vtab, const float* __restrict__ ctab,
                  const float* __restrict__ ptab, const float* __restrict__ cw,
                  const float* __restrict__ lnw, const float* __restrict__ lnb,
                  const __bf16* __restrict__ Tf, const __bf16* __restrict__ Ti,
                  float* __restrict__ out) {
    extern __shared__ char smem[];
    float*  xf = (float*)smem;               // [200][132] f32
    __bf16* xt = (__bf16*)(smem + XT_OFF);   // [128][224] bf16 (x^T, then G^T)
    float*  Fb = (float*)(smem + FB_OFF);    // [224][132] f32 (F, then h)

    const int r = blockIdx.x;                // 0..1279 = stream*256 + batch
    const int strm = r >> 8, batch = r & 255;
    const int tid = threadIdx.x, lane = tid & 31, wid = tid >> 5;

    // ---- gather one [200][128] slab from the right embedding table --------
    const float* tab;
    const int* ids = nullptr;
    if      (strm == 0) { tab = qtab; ids = qids; }
    else if (strm == 1) { tab = utab; ids = uids; }
    else if (strm == 2) { tab = vtab; ids = vids; }
    else if (strm == 3) { tab = ctab; ids = clicks; }
    else                { tab = ptab; }

    for (int i = tid; i < SEQ * EMB; i += 224) {
        const int s = i >> 7, e = i & 127;
        const int row = (strm == 4) ? (s % 10) : ids[batch * SEQ + s];
        const float v = tab[row * EMB + e];
        xf[s * FROW + e] = v;
        xt[e * SP + s]   = (__bf16)v;
    }
    for (int i = tid; i < EMB * (SP - SEQ); i += 224) {   // zero K padding
        const int e = i / 24, s = SEQ + (i % 24);
        xt[e * SP + s] = (__bf16)0.f;
    }
    __syncthreads();

    // ---- GEMM1: F = Tf @ x  -> Fb ----------------------------------------
    {
        v8f acc[2][8];
        const v8f z = {};
#pragma unroll
        for (int a = 0; a < 2; ++a)
#pragma unroll
            for (int b = 0; b < 8; ++b) acc[a][b] = z;
        run_gemm(Tf, xt, wid, lane, acc);
        // C/D layout: lane<16 -> N=lane, VGPR v -> M=v; lane>=16 -> M=v+8.
#pragma unroll
        for (int a = 0; a < 2; ++a)
#pragma unroll
            for (int e8 = 0; e8 < 8; ++e8)
#pragma unroll
                for (int v = 0; v < 8; ++v) {
                    const int mr = (wid + a * 7) * 16 + v + ((lane >> 4) << 3);
                    const int ec = e8 * 16 + (lane & 15);
                    Fb[mr * FROW + ec] = acc[a][e8][v];
                }
    }
    __syncthreads();

    // ---- pointwise complex filter: G = w .* F -> xt (bf16, transposed) ---
    for (int i = tid; i < NFREQ * EMB; i += 224) {
        const int k = i >> 7, e = i & 127;
        const float fr = Fb[k * FROW + e];
        const float fi = Fb[(112 + k) * FROW + e];
        const float wr = cw[(k * EMB + e) * 2 + 0];
        const float wi = cw[(k * EMB + e) * 2 + 1];
        xt[e * SP + k]       = (__bf16)(wr * fr - wi * fi);
        xt[e * SP + 112 + k] = (__bf16)(wr * fi + wi * fr);
    }
    for (int i = tid; i < EMB * 22; i += 224) {           // zero pad rows
        const int e = i / 22, j = i % 22;
        const int k = (j < 11) ? (101 + j) : (202 + j);   // 101..111, 213..223
        xt[e * SP + k] = (__bf16)0.f;
    }
    __syncthreads();

    // ---- GEMM2: Y = Ti @ G, fused residual h = y + x -> Fb ---------------
    v8f acc[2][8];
    {
        const v8f z = {};
#pragma unroll
        for (int a = 0; a < 2; ++a)
#pragma unroll
            for (int b = 0; b < 8; ++b) acc[a][b] = z;
    }
    run_gemm(Ti, xt, wid, lane, acc);
#pragma unroll
    for (int a = 0; a < 2; ++a)
#pragma unroll
        for (int e8 = 0; e8 < 8; ++e8)
#pragma unroll
            for (int v = 0; v < 8; ++v) {
                const int t = (wid + a * 7) * 16 + v + ((lane >> 4) << 3);
                if (t < SEQ) {
                    const int ec = e8 * 16 + (lane & 15);
                    Fb[t * FROW + ec] = acc[a][e8][v] + xf[t * FROW + ec];
                }
            }
    __syncthreads();

    // ---- LayerNorm over E (wave32 shfl reduction), write out -------------
    for (int t = wid; t < SEQ; t += 7) {
        const int e0 = lane * 4;
        float v0 = Fb[t * FROW + e0 + 0];
        float v1 = Fb[t * FROW + e0 + 1];
        float v2 = Fb[t * FROW + e0 + 2];
        float v3 = Fb[t * FROW + e0 + 3];
        float sum = v0 + v1 + v2 + v3;
#pragma unroll
        for (int off = 16; off; off >>= 1) sum += __shfl_xor(sum, off, 32);
        const float mu = sum * 0.0078125f;                // /128
        const float d0 = v0 - mu, d1 = v1 - mu, d2 = v2 - mu, d3 = v3 - mu;
        float s2 = d0 * d0 + d1 * d1 + d2 * d2 + d3 * d3;
#pragma unroll
        for (int off = 16; off; off >>= 1) s2 += __shfl_xor(s2, off, 32);
        const float rstd = rsqrtf(s2 * 0.0078125f + 1e-12f);
        const size_t base = ((size_t)r * SEQ + t) * EMB + e0;
        out[base + 0] = lnw[e0 + 0] * d0 * rstd + lnb[e0 + 0];
        out[base + 1] = lnw[e0 + 1] * d1 * rstd + lnb[e0 + 1];
        out[base + 2] = lnw[e0 + 2] * d2 * rstd + lnb[e0 + 2];
        out[base + 3] = lnw[e0 + 3] * d3 * rstd + lnb[e0 + 3];
    }
}

// ------------------------------- launcher ----------------------------------
extern "C" void kernel_launch(void* const* d_in, const int* in_sizes, int n_in,
                              void* d_out, int out_size, void* d_ws, size_t ws_size,
                              hipStream_t stream) {
    (void)in_sizes; (void)n_in; (void)out_size;
    if (ws_size < (size_t)(2 * SP * SP * 2)) return;   // need 200,704 B scratch

    const int*   qids = (const int*)d_in[0];
    const int*   uids = (const int*)d_in[1];
    const int*   vids = (const int*)d_in[2];
    const int*   clks = (const int*)d_in[3];
    const float* qtab = (const float*)d_in[4];
    const float* utab = (const float*)d_in[5];
    const float* vtab = (const float*)d_in[6];
    const float* ctab = (const float*)d_in[7];
    const float* ptab = (const float*)d_in[8];
    const float* cw   = (const float*)d_in[9];
    const float* lnw  = (const float*)d_in[10];
    const float* lnb  = (const float*)d_in[11];

    __bf16* Tf = (__bf16*)d_ws;
    __bf16* Ti = Tf + SP * SP;

    (void)hipFuncSetAttribute(reinterpret_cast<const void*>(filter_fused),
                              hipFuncAttributeMaxDynamicSharedMemorySize,
                              LDS_BYTES);

    // 2 * 224 * 224 = 100352 = 392 * 256 entries
    build_transforms<<<392, 256, 0, stream>>>(Tf, Ti);

    filter_fused<<<1280, 224, LDS_BYTES, stream>>>(
        qids, uids, vids, clks, qtab, utab, vtab, ctab, ptab,
        cw, lnw, lnb, Tf, Ti, (float*)d_out);
}
